// ContinuousREWAEncoder_84293028151335
// MI455X (gfx1250) — matrix operand order repaired and checked by reading, not compiled, not run on hardware
//
#include <hip/hip_runtime.h>
#include <hip/hip_bf16.h>
#include <math.h>

// ---------------------------------------------------------------------------
// ContinuousREWAEncoder for MI455X (gfx1250):
//   out = fwht(x @ W^T) + noise  ==  x @ (H W / sqrt(32))^T + noise
// FWHT folded into a tiny W-preprocessing pass; hot loop is a skinny GEMM
// (32768 x 1024) x (1024 x 32) on v_wmma_f32_16x16x32_bf16 (f32 accumulate).
// Memory-bound: 134 MB of f32 x at 23.3 TB/s ~= 5.9 us floor; per-iteration
// work is kept to 4 NT b128 loads + 8 v_perm packs + 4 ds_load_b128 + 2 WMMA.
// ---------------------------------------------------------------------------

typedef __attribute__((ext_vector_type(16))) __bf16        v16bf;
typedef __attribute__((ext_vector_type(8)))  float         v8f;
typedef __attribute__((ext_vector_type(4)))  float         f32x4;  // clang ext-vector
typedef __attribute__((ext_vector_type(4)))  unsigned int  u32x4;

#define D_DIM 1024
#define M_DIM 32
#define ROWS  32768           // B*N = 4*8192

// f32 -> bf16 pair, truncating, via one v_perm_b32: result = {hi[31:16], lo[31:16]}
__device__ __forceinline__ unsigned int pack_bf16_trunc(float lo, float hi) {
  return __builtin_amdgcn_perm(__float_as_uint(hi), __float_as_uint(lo), 0x07060302u);
}

// round-to-nearest-even f32 -> bf16 (used only in the one-shot W prep)
__device__ __forceinline__ unsigned short f32_to_bf16_rne(float f) {
  union { float f; unsigned int u; } v; v.f = f;
  unsigned int r = v.u + 0x7FFFu + ((v.u >> 16) & 1u);
  return (unsigned short)(r >> 16);
}

// deterministic per-element Gaussian (hash + Box-Muller); stands in for the
// fixed-key jax noise stream (exact threefry reproduction is impossible here).
__device__ __forceinline__ unsigned int hash_u32(unsigned int x) {
  x ^= x >> 16; x *= 0x7feb352du;
  x ^= x >> 15; x *= 0x846ca68bu;
  x ^= x >> 16; return x;
}
__device__ __forceinline__ float gauss01(unsigned int idx) {
  unsigned int h1 = hash_u32(idx * 2u + 0x9e3779b9u);
  unsigned int h2 = hash_u32(idx * 2u + 0x85ebca6bu);
  float u1 = ((float)(h1 >> 8) + 0.5f) * (1.0f / 16777216.0f);
  float u2 = (float)(h2 >> 8) * (1.0f / 16777216.0f);
  return __fsqrt_rn(-2.0f * __logf(u1)) * __cosf(6.28318530717958647f * u2);
}

// ---------------------------------------------------------------------------
// Prep: W'[n][k] = (1/sqrt(32)) * sum_m (-1)^popc(n&m) W[m][k], emitted as
// bf16 directly in the WMMA B-operand swizzle.
// ws layout: 64 blocks of (32 lanes x 16 bf16); block = kb*2 + half,
//   half selects columns n = half*16 + (lane&15),
//   lane group (lane>>4) selects K {0-7,16-23} vs {8-15,24-31} within the 32-wide K block,
//   element j in lane -> K = kb*32 + (lane>>4)*8 + (j&7) + (j>>3)*16   (2 bf16 per VGPR).
// ---------------------------------------------------------------------------
__global__ __launch_bounds__(256) void rewa_prep_w(const float* __restrict__ W,
                                                   unsigned short* __restrict__ wp) {
  int t = blockIdx.x * blockDim.x + threadIdx.x;   // 32768 threads total
  int j    =  t        & 15;
  int lane = (t >> 4)  & 31;
  int half = (t >> 9)  & 1;
  int kb   =  t >> 10;                             // 0..31
  int n    = half * 16 + (lane & 15);
  int k    = kb * 32 + ((lane >> 4) * 8) + (j & 7) + ((j >> 3) * 16);
  float s = 0.0f;
#pragma unroll
  for (int m = 0; m < 32; ++m) {
    float w = W[m * D_DIM + k];
    s += (__popc(n & m) & 1) ? -w : w;
  }
  s *= 0.17677669529663687f;                       // 1/sqrt(32)
  wp[t] = f32_to_bf16_rne(s);
}

// ---------------------------------------------------------------------------
// Main GEMM: one wave -> 16 output rows x all 32 columns.
// ---------------------------------------------------------------------------
__global__ __launch_bounds__(256) void rewa_gemm(const float* __restrict__ x,
                                                 const unsigned short* __restrict__ wp,
                                                 float* __restrict__ out) {
  __shared__ unsigned short bsm[64 * 512];         // 64 KB: full swizzled W'

  // cooperative stage of W' into LDS (4096 x b128)
  {
    const u32x4* src = (const u32x4*)wp;
    u32x4*       dst = (u32x4*)bsm;
#pragma unroll
    for (int i = 0; i < 16; ++i)
      dst[threadIdx.x + 256 * i] = src[threadIdx.x + 256 * i];
  }
  __syncthreads();

  const int lane    = threadIdx.x & 31;
  const int waveId  = threadIdx.x >> 5;
  const int tile    = blockIdx.x * 8 + waveId;     // 0..2047
  const int rowbase = tile * 16;
  const int l       = lane & 15;
  const int halfSel = lane >> 4;                   // A-operand K-group select

  // A layout (ISA 16-bit A 16x32): lanes 0-15 -> K {0..7,16..23},
  // lanes 16-31 -> K {8..15,24..31}; elem j<8 -> K=base+j, j>=8 -> K=base+16+(j-8)
  const float* xrow = x + (size_t)(rowbase + l) * D_DIM + halfSel * 8;

  v8f acc0 = {0.f,0.f,0.f,0.f,0.f,0.f,0.f,0.f};
  v8f acc1 = {0.f,0.f,0.f,0.f,0.f,0.f,0.f,0.f};

  union AU { unsigned int u[8]; v16bf v; };
  union BU { u32x4 q[2];        v16bf v; };

#pragma unroll 2
  for (int kb = 0; kb < 32; ++kb) {
    const float* p = xrow + kb * 32;
    __builtin_prefetch(p + 128, 0, 1);             // global_prefetch_b8, 4 iters ahead

    // streaming, single-use: non-temporal b128 loads
    f32x4 a0 = __builtin_nontemporal_load((const f32x4*)(p +  0));
    f32x4 a1 = __builtin_nontemporal_load((const f32x4*)(p +  4));
    f32x4 a2 = __builtin_nontemporal_load((const f32x4*)(p + 16));
    f32x4 a3 = __builtin_nontemporal_load((const f32x4*)(p + 20));

    AU a;                                          // 8x v_perm_b32 total
    a.u[0] = pack_bf16_trunc(a0.x, a0.y);
    a.u[1] = pack_bf16_trunc(a0.z, a0.w);
    a.u[2] = pack_bf16_trunc(a1.x, a1.y);
    a.u[3] = pack_bf16_trunc(a1.z, a1.w);
    a.u[4] = pack_bf16_trunc(a2.x, a2.y);
    a.u[5] = pack_bf16_trunc(a2.z, a2.w);
    a.u[6] = pack_bf16_trunc(a3.x, a3.y);
    a.u[7] = pack_bf16_trunc(a3.z, a3.w);

    const u32x4* b0p = (const u32x4*)(bsm + ((kb * 2 + 0) * 32 + lane) * 16);
    const u32x4* b1p = (const u32x4*)(bsm + ((kb * 2 + 1) * 32 + lane) * 16);
    BU b0, b1;
    b0.q[0] = b0p[0]; b0.q[1] = b0p[1];            // ds_load_b128 x2
    b1.q[0] = b1p[0]; b1.q[1] = b1p[1];

    acc0 = __builtin_amdgcn_wmma_f32_16x16x32_bf16(false, a.v, false, b0.v,
                                                   (short)0, acc0, false, false);
    acc1 = __builtin_amdgcn_wmma_f32_16x16x32_bf16(false, a.v, false, b1.v,
                                                   (short)0, acc1, false, false);
  }

  // C/D layout: VGPR v, lanes 0-15 -> row v,   N = lane
  //                     lanes 16-31 -> row v+8, N = lane-16
#pragma unroll
  for (int v = 0; v < 8; ++v) {
    const int r = rowbase + v + halfSel * 8;
    const unsigned int i0 = (unsigned int)r * M_DIM + l;        // acc0: n = 0..15
    const unsigned int i1 = (unsigned int)r * M_DIM + 16 + l;   // acc1: n = 16..31
    __builtin_nontemporal_store(acc0[v] + 0.01f * gauss01(i0), out + i0);
    __builtin_nontemporal_store(acc1[v] + 0.01f * gauss01(i1), out + i1);
  }
}

// ---------------------------------------------------------------------------
extern "C" void kernel_launch(void* const* d_in, const int* in_sizes, int n_in,
                              void* d_out, int out_size, void* d_ws, size_t ws_size,
                              hipStream_t stream) {
  const float* x = (const float*)d_in[0];          // (4, 8192, 1024) f32
  const float* W = (const float*)d_in[1];          // (32, 1024) f32
  float* out = (float*)d_out;                      // (4, 8192, 32) f32
  unsigned short* wp = (unsigned short*)d_ws;      // 64 KB swizzled bf16 W'

  rewa_prep_w<<<128, 256, 0, stream>>>(W, wp);     // 32768 threads
  rewa_gemm<<<256, 256, 0, stream>>>(x, wp, out);  // 2048 waves, 1 tile each
}
